// Generator_2705829397359
// MI455X (gfx1250) — 8-bit matrix operand order repaired and checked by reading, compiled, not run on hardware
//
#include <hip/hip_runtime.h>
#include <stdint.h>
#include <math.h>

typedef __attribute__((ext_vector_type(8))) int v8i;

#define K_DIM 1024
#define QMAXF 7.0f

__global__ void init_scales_kernel(unsigned* absbits) {
    if (threadIdx.x < 2) absbits[threadIdx.x] = 0u;
}

__global__ __launch_bounds__(256)
void absmax_kernel(const float* __restrict__ p, long n, unsigned* __restrict__ out) {
    __shared__ unsigned sm[256];
    unsigned m = 0u;
    long stride = (long)gridDim.x * blockDim.x;
    for (long i = (long)blockIdx.x * blockDim.x + threadIdx.x; i < n; i += stride) {
        unsigned b = __float_as_uint(fabsf(p[i]));
        m = (b > m) ? b : m;
    }
    sm[threadIdx.x] = m;
    __syncthreads();
    for (int s = 128; s > 0; s >>= 1) {
        if (threadIdx.x < s) {
            unsigned o = sm[threadIdx.x + s];
            if (o > sm[threadIdx.x]) sm[threadIdx.x] = o;
        }
        __syncthreads();
    }
    if (threadIdx.x == 0) atomicMax(out, sm[0]);
}

__global__ __launch_bounds__(256)
void quantize_kernel(const float* __restrict__ p, long n,
                     const unsigned* __restrict__ absbits, int which,
                     int8_t* __restrict__ q) {
    long i = (long)blockIdx.x * blockDim.x + threadIdx.x;
    if (i >= n) return;
    float s = __uint_as_float(absbits[which]) / QMAXF;
    int qi = (int)rintf(p[i] / s);
    qi = qi > 7 ? 7 : (qi < -8 ? -8 : qi);
    q[i] = (int8_t)qi;
}

__global__ __launch_bounds__(256)
void quantize_bias_kernel(const float* __restrict__ b, int vocab,
                          const unsigned* __restrict__ absbits,
                          float* __restrict__ bq) {
    int i = blockIdx.x * blockDim.x + threadIdx.x;
    if (i >= vocab) return;
    float sx = __uint_as_float(absbits[0]) / QMAXF;
    float sw = __uint_as_float(absbits[1]) / QMAXF;
    float sb = sx * sw;
    bq[i] = rintf(b[i] / sb) * sb;
}

__global__ __launch_bounds__(256)
void gemm_i8_kernel(const int8_t* __restrict__ xq, const int8_t* __restrict__ wq,
                    const float* __restrict__ bq, const unsigned* __restrict__ absbits,
                    float* __restrict__ out, int vocab) {
    const int lane = threadIdx.x & 31;
    const int wave = threadIdx.x >> 5;
    const int m0   = blockIdx.y * 32;
    const int nChunk = blockIdx.x * 8 + wave;
    const int n0   = nChunk * 64;
    if (n0 >= vocab) return;

    const int half = lane >> 4;
    const int l16  = lane & 15;

    v8i acc0[4] = {};
    v8i acc1[4] = {};

    const int8_t* arow0 = xq + (size_t)(m0 + l16) * K_DIM + half * 16;
    const int8_t* arow1 = arow0 + (size_t)16 * K_DIM;

    int  ncol[4];
    bool vld[4];
    const int8_t* brow[4];
#pragma unroll
    for (int t = 0; t < 4; ++t) {
        ncol[t] = n0 + t * 16 + l16;
        vld[t]  = (ncol[t] < vocab);
        int nc  = vld[t] ? ncol[t] : (vocab - 1);
        brow[t] = wq + (size_t)nc * K_DIM + half * 16;
    }

#pragma unroll 4
    for (int k0 = 0; k0 < K_DIM; k0 += 64) {
        int2 p0 = *(const int2*)(arow0 + k0);
        int2 p1 = *(const int2*)(arow0 + k0 + 8);
        int2 p2 = *(const int2*)(arow0 + k0 + 32);
        int2 p3 = *(const int2*)(arow0 + k0 + 40);
        v8i a0 = {p0.x, p0.y, p1.x, p1.y, p2.x, p2.y, p3.x, p3.y};
        int2 q0 = *(const int2*)(arow1 + k0);
        int2 q1 = *(const int2*)(arow1 + k0 + 8);
        int2 q2 = *(const int2*)(arow1 + k0 + 32);
        int2 q3 = *(const int2*)(arow1 + k0 + 40);
        v8i a1 = {q0.x, q0.y, q1.x, q1.y, q2.x, q2.y, q3.x, q3.y};

#pragma unroll
        for (int t = 0; t < 4; ++t) {
            int4 lo = *(const int4*)(brow[t] + k0);
            int4 hi = *(const int4*)(brow[t] + k0 + 32);
            v8i bf = {lo.x, lo.y, lo.z, lo.w, hi.x, hi.y, hi.z, hi.w};
            acc0[t] = __builtin_amdgcn_wmma_i32_16x16x64_iu8(
                true, a0, true, bf, acc0[t], false, false);
            acc1[t] = __builtin_amdgcn_wmma_i32_16x16x64_iu8(
                true, a1, true, bf, acc1[t], false, false);
        }
    }

    const float sx = __uint_as_float(absbits[0]) / QMAXF;
    const float sw = __uint_as_float(absbits[1]) / QMAXF;
    const float sb = sx * sw;
    const int mrowA = m0 + half * 8;
    const int mrowB = mrowA + 16;
#pragma unroll
    for (int t = 0; t < 4; ++t) {
        if (!vld[t]) continue;
        float bias = bq[ncol[t]];
#pragma unroll
        for (int r = 0; r < 8; ++r) {
            out[(size_t)(mrowA + r) * vocab + ncol[t]] = (float)acc0[t][r] * sb + bias;
        }
#pragma unroll
        for (int r = 0; r < 8; ++r) {
            out[(size_t)(mrowB + r) * vocab + ncol[t]] = (float)acc1[t][r] * sb + bias;
        }
    }
}

__global__ __launch_bounds__(256)
void rowlse_kernel(const float* __restrict__ logits, int vocab, float* __restrict__ lse) {
    __shared__ float sm[256];
    __shared__ float ss[256];
    const int row = blockIdx.x;
    const int tid = threadIdx.x;
    const float* p = logits + (size_t)row * vocab;
    float m = -INFINITY, s = 0.0f;
    for (int i = tid; i < vocab; i += 256) {
        float v = p[i];
        if (v > m) { s = s * expf(m - v) + 1.0f; m = v; }
        else       { s += expf(v - m); }
    }
    sm[tid] = m; ss[tid] = s;
    __syncthreads();
    for (int k = 128; k > 0; k >>= 1) {
        if (tid < k) {
            float m2 = sm[tid + k], s2 = ss[tid + k];
            float M = fmaxf(sm[tid], m2);
            ss[tid] = ss[tid] * expf(sm[tid] - M) + s2 * expf(m2 - M);
            sm[tid] = M;
        }
        __syncthreads();
    }
    if (tid == 0) lse[row] = sm[0] + logf(ss[0]);
}

__global__ __launch_bounds__(256)
void sub_lse_kernel(float* __restrict__ out, const float* __restrict__ lse, int vocab) {
    int col = blockIdx.x * blockDim.x + threadIdx.x;
    int row = blockIdx.y;
    if (col < vocab) out[(size_t)row * vocab + col] -= lse[row];
}

static inline size_t align256(size_t v) { return (v + 255) & ~(size_t)255; }

extern "C" void kernel_launch(void* const* d_in, const int* in_sizes, int n_in,
                              void* d_out, int out_size, void* d_ws, size_t ws_size,
                              hipStream_t stream) {
    const float* x = (const float*)d_in[0];
    const float* W = (const float*)d_in[1];
    const float* b = (const float*)d_in[2];
    const long nx = in_sizes[0];
    const long nw = in_sizes[1];
    const int vocab = in_sizes[2];
    const int M = (int)(nx / K_DIM);
    float* out = (float*)d_out;

    char* ws = (char*)d_ws;
    size_t off = 0;
    unsigned* absbits = (unsigned*)(ws + off); off = align256(off + 2 * sizeof(unsigned));
    float* bq  = (float*)(ws + off);           off = align256(off + (size_t)vocab * sizeof(float));
    float* lse = (float*)(ws + off);           off = align256(off + (size_t)M * sizeof(float));
    int8_t* xq = (int8_t*)(ws + off);          off = align256(off + (size_t)nx);
    int8_t* wq = (int8_t*)(ws + off);          off = align256(off + (size_t)nw);
    (void)ws_size;

    init_scales_kernel<<<1, 32, 0, stream>>>(absbits);
    absmax_kernel<<<1024, 256, 0, stream>>>(x, nx, absbits + 0);
    absmax_kernel<<<4096, 256, 0, stream>>>(W, nw, absbits + 1);

    quantize_kernel<<<(unsigned)((nx + 255) / 256), 256, 0, stream>>>(x, nx, absbits, 0, xq);
    quantize_kernel<<<(unsigned)((nw + 255) / 256), 256, 0, stream>>>(W, nw, absbits, 1, wq);
    quantize_bias_kernel<<<(vocab + 255) / 256, 256, 0, stream>>>(b, vocab, absbits, bq);

    int nTiles64 = (vocab + 63) / 64;
    dim3 gGrid((nTiles64 + 7) / 8, M / 32);
    gemm_i8_kernel<<<gGrid, 256, 0, stream>>>(xq, wq, bq, absbits, out, vocab);

    rowlse_kernel<<<M, 256, 0, stream>>>(out, vocab, lse);
    dim3 sGrid((vocab + 255) / 256, M);
    sub_lse_kernel<<<sGrid, 256, 0, stream>>>(out, lse, vocab);
}
